// LSTMSeq2SeqDifferent_65051574665798
// MI455X (gfx1250) — matrix operand order repaired and checked by reading, not compiled, not run on hardware
//
#include <hip/hip_runtime.h>
#include <hip/hip_bf16.h>
#include <math.h>

typedef __attribute__((ext_vector_type(16))) __bf16 v16bf;
typedef __attribute__((ext_vector_type(8)))  float  v8f;

union FragBF { v16bf v; uint4 q[2]; unsigned short u[16]; };

#define T_STEPS 384
#define BATCH   128
#define INDIM   512
#define HIDDIM  1024
#define OUTDIM  512
#define KTOT    (INDIM + HIDDIM)   // 1536
#define FOURH   (4 * HIDDIM)       // 4096
#define KP      40                 // padded LDS k-stride in ushorts (80B, 16B-aligned)

__device__ __forceinline__ unsigned short f2bf(float f) {
  unsigned u = __float_as_uint(f);
  return (unsigned short)((u + 0x7FFFu + ((u >> 16) & 1u)) >> 16);
}
__device__ __forceinline__ float sigmoidf(float x) { return 1.0f / (1.0f + __expf(-x)); }

// ---------------- elementwise helpers ----------------
__global__ void zero_f32(float* __restrict__ p, int n) {
  int i = blockIdx.x * blockDim.x + threadIdx.x;
  if (i < n) p[i] = 0.0f;
}

// ---------------- one-time transpose + fp32->bf16 convert ----------------
// src: [K, N] fp32 row-major; dst: [N, dstStride] bf16, written at column offset colOff.
// grid: (K/32, N/32), block 256 (32x8).
__global__ __launch_bounds__(256) void cvt_transpose_bf16(
    const float* __restrict__ src, unsigned short* __restrict__ dst,
    int K, int N, int dstStride, int colOff)
{
  __shared__ float tile[32][33];
  const int kb = blockIdx.x * 32, nb = blockIdx.y * 32;
  const int tx = threadIdx.x & 31, ty = threadIdx.x >> 5;   // ty: 0..7
  #pragma unroll
  for (int r = 0; r < 32; r += 8)
    tile[ty + r][tx] = src[(size_t)(kb + ty + r) * N + nb + tx];
  __syncthreads();
  #pragma unroll
  for (int r = 0; r < 32; r += 8)
    dst[(size_t)(nb + ty + r) * dstStride + colOff + kb + tx] = f2bf(tile[tx][ty + r]);
}

// ---------------- fused LSTM step: gates = [x|h] @ W + b, then cell ----------------
// Wt is pre-transposed: [FOURH rows n][KTOT cols k] bf16 (n = g*HID + j).
// grid: (BATCH/64, HIDDIM/64), block: 256 (8 waves of 32)
__global__ __launch_bounds__(256) void lstm_step(
    const float* __restrict__ xin,            // [BATCH, INDIM]  fp32
    const float* __restrict__ hin,            // [BATCH, HIDDIM] fp32
    const unsigned short* __restrict__ Wt,    // [FOURH, KTOT] bf16 (transposed)
    const float* __restrict__ bias,           // [FOURH]
    float* __restrict__ cbuf,                 // [BATCH, HIDDIM] in/out
    float* __restrict__ hout)                 // [BATCH, HIDDIM]
{
  __shared__ unsigned short aLds[64 * KP];
  __shared__ unsigned short bLds[4 * 64 * KP];

  const int tid  = threadIdx.x;
  const int lane = tid & 31;
  const int wave = tid >> 5;
  const int mw   = wave >> 1;                 // 0..3 : M sub-tile
  const int nw   = wave & 1;                  // 0..1 : N half
  const int m0   = blockIdx.x * 64;
  const int j0   = blockIdx.y * 64;

  v8f zero{};
  v8f acc[4][2];
  for (int g = 0; g < 4; ++g) for (int ns = 0; ns < 2; ++ns) acc[g][ns] = zero;

  for (int kc = 0; kc < KTOT / 32; ++kc) {
    const int kbase = kc * 32;
    if (kbase + 32 < KTOT)                    // global_prefetch_b8 of next K slab
      __builtin_prefetch(&Wt[(size_t)j0 * KTOT + kbase + 32], 0, 1);

    __syncthreads();
    // stage A: 64 x 32 fp32 -> bf16, float4 reads + ds_store_b64
    #pragma unroll
    for (int p = 0; p < 2; ++p) {
      int q4 = tid + p * 256;                 // 0..511 quads
      int m  = q4 >> 3;
      int k4 = (q4 & 7) * 4;
      int kg = kbase + k4;
      float4 v = (kg < INDIM)
          ? *(const float4*)&xin[(size_t)(m0 + m) * INDIM + kg]
          : *(const float4*)&hin[(size_t)(m0 + m) * HIDDIM + (kg - INDIM)];
      ushort4 s;
      s.x = f2bf(v.x); s.y = f2bf(v.y); s.z = f2bf(v.z); s.w = f2bf(v.w);
      *(ushort4*)&aLds[m * KP + k4] = s;
    }
    // stage B: 4 gates x 64 j x 32 k, uint4 reads from pre-transposed Wt + ds_store_b128
    #pragma unroll
    for (int p = 0; p < 4; ++p) {
      int q16 = tid + p * 256;                // 0..1023 chunks of 8 ushorts
      int k8  = (q16 & 3) * 8;
      int j   = (q16 >> 2) & 63;
      int g   = q16 >> 8;                     // 0..3
      uint4 w = *(const uint4*)&Wt[(size_t)(g * HIDDIM + j0 + j) * KTOT + kbase + k8];
      *(uint4*)&bLds[(g * 64 + j) * KP + k8] = w;
    }
    __syncthreads();

    // A fragment: lane L -> row (L&15); K halves {koff..+7, koff+16..+23}, koff = (L>=16)?8:0
    FragBF a;
    {
      int row  = mw * 16 + (lane & 15);
      int koff = (lane >= 16) ? 8 : 0;
      a.q[0] = *(const uint4*)&aLds[row * KP + koff];
      a.q[1] = *(const uint4*)&aLds[row * KP + koff + 16];
    }
    #pragma unroll
    for (int g = 0; g < 4; ++g) {
      #pragma unroll
      for (int ns = 0; ns < 2; ++ns) {
        FragBF b;                             // lane L -> col (L&15); K halves {koff..+7, koff+16..+23}, koff = (L>=16)?8:0
        int jl   = nw * 32 + ns * 16 + (lane & 15);
        int koff = (lane >= 16) ? 8 : 0;
        b.q[0] = *(const uint4*)&bLds[(g * 64 + jl) * KP + koff];
        b.q[1] = *(const uint4*)&bLds[(g * 64 + jl) * KP + koff + 16];
        acc[g][ns] = __builtin_amdgcn_wmma_f32_16x16x32_bf16(
            false, a.v, false, b.v, (short)0, acc[g][ns], false, false);
      }
    }
  }

  // epilogue: LSTM cell (elementwise on matching accumulator slots)
  #pragma unroll
  for (int ns = 0; ns < 2; ++ns) {
    int j = j0 + nw * 32 + ns * 16 + (lane & 15);
    float bi = bias[0 * HIDDIM + j];
    float bf = bias[1 * HIDDIM + j];
    float bg = bias[2 * HIDDIM + j];
    float bo = bias[3 * HIDDIM + j];
    #pragma unroll
    for (int r = 0; r < 8; ++r) {
      int m = m0 + mw * 16 + r + ((lane >= 16) ? 8 : 0);
      float ig = sigmoidf(acc[0][ns][r] + bi);
      float fg = sigmoidf(acc[1][ns][r] + bf);
      float gg = tanhf(acc[2][ns][r] + bg);
      float og = sigmoidf(acc[3][ns][r] + bo);
      size_t idx = (size_t)m * HIDDIM + j;
      float cn = fg * cbuf[idx] + ig * gg;
      cbuf[idx] = cn;
      hout[idx] = og * tanhf(cn);
    }
  }
}

// ---------------- fc GEMM: logits = h @ fcW  (bias+softmax separate) ----------------
// WfT pre-transposed: [OUTDIM rows n][HIDDIM cols k]. grid: (BATCH/64, OUTDIM/64), block 256
__global__ __launch_bounds__(256) void fc_gemm(
    const float* __restrict__ hin,            // [BATCH, HIDDIM]
    const unsigned short* __restrict__ WfT,   // [OUTDIM, HIDDIM] bf16
    float* __restrict__ logits)               // [BATCH, OUTDIM]
{
  __shared__ unsigned short aLds[64 * KP];
  __shared__ unsigned short bLds[64 * KP];

  const int tid  = threadIdx.x;
  const int lane = tid & 31;
  const int wave = tid >> 5;
  const int mw   = wave >> 1;
  const int nw   = wave & 1;
  const int m0   = blockIdx.x * 64;
  const int j0   = blockIdx.y * 64;

  v8f zero{};
  v8f acc[2] = {zero, zero};

  for (int kc = 0; kc < HIDDIM / 32; ++kc) {
    const int kbase = kc * 32;
    if (kbase + 32 < HIDDIM)
      __builtin_prefetch(&WfT[(size_t)j0 * HIDDIM + kbase + 32], 0, 1);

    __syncthreads();
    #pragma unroll
    for (int p = 0; p < 2; ++p) {
      int q4 = tid + p * 256;
      int m  = q4 >> 3;
      int k4 = (q4 & 7) * 4;
      float4 v = *(const float4*)&hin[(size_t)(m0 + m) * HIDDIM + kbase + k4];
      ushort4 s;
      s.x = f2bf(v.x); s.y = f2bf(v.y); s.z = f2bf(v.z); s.w = f2bf(v.w);
      *(ushort4*)&aLds[m * KP + k4] = s;
    }
    {
      int k8 = (tid & 3) * 8;
      int j  = tid >> 2;                      // 0..63
      uint4 w = *(const uint4*)&WfT[(size_t)(j0 + j) * HIDDIM + kbase + k8];
      *(uint4*)&bLds[j * KP + k8] = w;
    }
    __syncthreads();

    FragBF a;
    {
      int row  = mw * 16 + (lane & 15);
      int koff = (lane >= 16) ? 8 : 0;
      a.q[0] = *(const uint4*)&aLds[row * KP + koff];
      a.q[1] = *(const uint4*)&aLds[row * KP + koff + 16];
    }
    #pragma unroll
    for (int ns = 0; ns < 2; ++ns) {
      FragBF b;
      int jl   = nw * 32 + ns * 16 + (lane & 15);
      int koff = (lane >= 16) ? 8 : 0;
      b.q[0] = *(const uint4*)&bLds[jl * KP + koff];
      b.q[1] = *(const uint4*)&bLds[jl * KP + koff + 16];
      acc[ns] = __builtin_amdgcn_wmma_f32_16x16x32_bf16(
          false, a.v, false, b.v, (short)0, acc[ns], false, false);
    }
  }

  #pragma unroll
  for (int ns = 0; ns < 2; ++ns) {
    int j = j0 + nw * 32 + ns * 16 + (lane & 15);
    #pragma unroll
    for (int r = 0; r < 8; ++r) {
      int m = m0 + mw * 16 + r + ((lane >= 16) ? 8 : 0);
      logits[(size_t)m * OUTDIM + j] = acc[ns][r];
    }
  }
}

// ---------------- row softmax over OUTDIM=512; writes next decoder input and output ----------------
__global__ __launch_bounds__(256) void softmax_row(
    const float* __restrict__ logits, const float* __restrict__ fcb,
    float* __restrict__ ynext, float* __restrict__ out)
{
  const int m = blockIdx.x, tid = threadIdx.x;
  __shared__ float red[256];
  float v0 = logits[(size_t)m * OUTDIM + tid]       + fcb[tid];
  float v1 = logits[(size_t)m * OUTDIM + tid + 256] + fcb[tid + 256];
  red[tid] = fmaxf(v0, v1);
  __syncthreads();
  for (int s = 128; s > 0; s >>= 1) { if (tid < s) red[tid] = fmaxf(red[tid], red[tid + s]); __syncthreads(); }
  float mx = red[0];
  __syncthreads();
  float e0 = __expf(v0 - mx), e1 = __expf(v1 - mx);
  red[tid] = e0 + e1;
  __syncthreads();
  for (int s = 128; s > 0; s >>= 1) { if (tid < s) red[tid] += red[tid + s]; __syncthreads(); }
  float inv = 1.0f / red[0];
  ynext[(size_t)m * OUTDIM + tid]       = e0 * inv;
  ynext[(size_t)m * OUTDIM + tid + 256] = e1 * inv;
  out[(size_t)m * OUTDIM + tid]         = e0 * inv;
  out[(size_t)m * OUTDIM + tid + 256]   = e1 * inv;
}

extern "C" void kernel_launch(void* const* d_in, const int* in_sizes, int n_in,
                              void* d_out, int out_size, void* d_ws, size_t ws_size,
                              hipStream_t stream) {
  (void)in_sizes; (void)n_in; (void)out_size; (void)ws_size;
  const float* x     = (const float*)d_in[0];
  const float* h0    = (const float*)d_in[1];
  const float* c0    = (const float*)d_in[2];
  const float* encWx = (const float*)d_in[3];
  const float* encWh = (const float*)d_in[4];
  const float* encB  = (const float*)d_in[5];
  const float* decWx = (const float*)d_in[6];
  const float* decWh = (const float*)d_in[7];
  const float* decB  = (const float*)d_in[8];
  const float* fcW   = (const float*)d_in[9];
  const float* fcB   = (const float*)d_in[10];
  float* out = (float*)d_out;

  char* ws = (char*)d_ws;
  size_t off = 0;
  auto carve = [&](size_t bytes) { void* p = ws + off; off += (bytes + 255) & ~(size_t)255; return p; };
  unsigned short* We = (unsigned short*)carve((size_t)FOURH * KTOT * 2);    // 12 MB, [4096][1536]
  unsigned short* Wd = (unsigned short*)carve((size_t)FOURH * KTOT * 2);    // 12 MB
  unsigned short* Wf = (unsigned short*)carve((size_t)OUTDIM * HIDDIM * 2); // 1 MB, [512][1024]
  float* hbuf0  = (float*)carve((size_t)BATCH * HIDDIM * 4);
  float* hbuf1  = (float*)carve((size_t)BATCH * HIDDIM * 4);
  float* cb     = (float*)carve((size_t)BATCH * HIDDIM * 4);
  float* ybuf0  = (float*)carve((size_t)BATCH * OUTDIM * 4);
  float* ybuf1  = (float*)carve((size_t)BATCH * OUTDIM * 4);
  float* logits = (float*)carve((size_t)BATCH * OUTDIM * 4);
  float* hbuf[2] = { hbuf0, hbuf1 };
  float* ybuf[2] = { ybuf0, ybuf1 };

  const int CT = 256;
  auto nb = [](int n, int b) { return (n + b - 1) / b; };

  // --- one-time weight transpose+convert into WMMA-friendly [N][K] bf16 layout ---
  cvt_transpose_bf16<<<dim3(INDIM / 32,  FOURH / 32), CT, 0, stream>>>(encWx, We, INDIM,  FOURH, KTOT, 0);
  cvt_transpose_bf16<<<dim3(HIDDIM / 32, FOURH / 32), CT, 0, stream>>>(encWh, We, HIDDIM, FOURH, KTOT, INDIM);
  cvt_transpose_bf16<<<dim3(OUTDIM / 32, FOURH / 32), CT, 0, stream>>>(decWx, Wd, OUTDIM, FOURH, KTOT, 0);
  cvt_transpose_bf16<<<dim3(HIDDIM / 32, FOURH / 32), CT, 0, stream>>>(decWh, Wd, HIDDIM, FOURH, KTOT, OUTDIM);
  cvt_transpose_bf16<<<dim3(HIDDIM / 32, OUTDIM / 32), CT, 0, stream>>>(fcW, Wf, HIDDIM, OUTDIM, HIDDIM, 0);

  // --- initial state ---
  hipMemcpyAsync(hbuf[0], h0, (size_t)BATCH * HIDDIM * 4, hipMemcpyDeviceToDevice, stream);
  hipMemcpyAsync(cb,      c0, (size_t)BATCH * HIDDIM * 4, hipMemcpyDeviceToDevice, stream);

  dim3 gStep(BATCH / 64, HIDDIM / 64);   // (2,16)
  dim3 gFc(BATCH / 64, OUTDIM / 64);     // (2,8)
  int hcur = 0;

  // --- encoder ---
  for (int t = 0; t < T_STEPS; ++t) {
    lstm_step<<<gStep, 256, 0, stream>>>(x + (size_t)t * BATCH * INDIM,
                                         hbuf[hcur], We, encB, cb, hbuf[hcur ^ 1]);
    hcur ^= 1;
  }

  // --- decoder init: c = 0, y0 = 0 ---
  zero_f32<<<nb(BATCH * HIDDIM, CT), CT, 0, stream>>>(cb, BATCH * HIDDIM);
  zero_f32<<<nb(BATCH * OUTDIM, CT), CT, 0, stream>>>(ybuf[0], BATCH * OUTDIM);
  int ycur = 0;

  // --- decoder ---
  for (int t = 0; t < T_STEPS; ++t) {
    lstm_step<<<gStep, 256, 0, stream>>>(ybuf[ycur], hbuf[hcur], Wd, decB, cb, hbuf[hcur ^ 1]);
    hcur ^= 1;
    fc_gemm<<<gFc, 256, 0, stream>>>(hbuf[hcur], Wf, logits);
    softmax_row<<<BATCH, 256, 0, stream>>>(logits, fcB, ybuf[ycur ^ 1],
                                           out + (size_t)t * BATCH * OUTDIM);
    ycur ^= 1;
  }
}
